// TCPGen_34978213659206
// MI455X (gfx1250) — matrix operand order repaired and checked by reading, not compiled, not run on hardware
//
#include <hip/hip_runtime.h>

// Problem constants (match reference)
#define BB   256
#define VV   50257
#define DD   256
#define EE   64
#define KK   200
#define HH   32
#define FFD  (DD + EE)          // 320
#define THRESH 0.5f
#define EPSV   1e-8f

typedef __attribute__((ext_vector_type(16))) _Float16 v16h;
typedef __attribute__((ext_vector_type(8)))  float    v8f;

// ---------------------------------------------------------------------------
// Kernel 1: pointer MLP scores via WMMA.
//   X[B*K, 320] = concat(decoder_state[b], bias_emb[subidx[b,k]])
//   H = relu(X @ W1 + b1); scores = H @ W2 + b2
// One wave -> 16 rows x 32 cols (two 16x16 f32 accumulators), K-loop of 10.
// ---------------------------------------------------------------------------
__global__ __launch_bounds__(256)
void mlp_scores_wmma(const float* __restrict__ dec,      // [B,D]
                     const float* __restrict__ bemb,     // [NSUB,E]
                     const float* __restrict__ W1,       // [F,H] row-major
                     const float* __restrict__ b1,       // [H]
                     const float* __restrict__ W2,       // [H]
                     const float* __restrict__ b2p,      // [1]
                     const int*   __restrict__ subidx,   // [B,K]
                     float* __restrict__ scores)         // [B*K]
{
    const int wave = threadIdx.x >> 5;
    const int lane = threadIdx.x & 31;
    const int gw   = blockIdx.x * (blockDim.x >> 5) + wave;
    const int R0   = gw * 16;            // 16 rows per wave

    const int m  = lane & 15;            // A-matrix row within tile
    const int hi = lane >> 4;            // half-wave group
    const int n  = lane & 15;            // B/C/D column

    const int row = R0 + m;              // flat (b,k)
    const int b   = row / KK;
    const int k   = row - b * KK;
    const float* __restrict__ drow = dec  + (long)b * DD;
    const int sub = subidx[b * KK + k];
    const float* __restrict__ erow = bemb + (long)sub * EE;

    v8f c0 = {};   // cols 0..15
    v8f c1 = {};   // cols 16..31

    #pragma unroll 1
    for (int kb = 0; kb < FFD / 32; ++kb) {
        const int F0 = kb * 32;

        // A fragment: 16x32 f16 (layout: K = hi*16 + e, matching B)
        v16h a;
        #pragma unroll
        for (int e = 0; e < 16; ++e) {
            const int kk = (hi << 4) + e;
            const int f  = F0 + kk;
            const float x = (f < DD) ? drow[f] : erow[f - DD];
            a[e] = (_Float16)x;
        }

        // B fragments: 32x16 f16 (layout as A: K = hi*16 + e, N = lane&15)
        v16h bt0, bt1;
        #pragma unroll
        for (int e = 0; e < 16; ++e) {
            const int kr = (hi << 4) + e;
            bt0[e] = (_Float16)W1[(F0 + kr) * HH + n];
            bt1[e] = (_Float16)W1[(F0 + kr) * HH + n + 16];
        }

        c0 = __builtin_amdgcn_wmma_f32_16x16x32_f16(false, a, false, bt0,
                                                    (short)0, c0, false, false);
        c1 = __builtin_amdgcn_wmma_f32_16x16x32_f16(false, a, false, bt1,
                                                    (short)0, c1, false, false);
    }

    // Epilogue: score[m] = sum_n relu(H[m,n]+b1[n]) * W2[n] + b2
    // D layout: lane holds (M = hi*8 + j, N = n) for j = 0..7.
    const float b1lo = b1[n],      b1hi = b1[n + 16];
    const float w2lo = W2[n],      w2hi = W2[n + 16];
    float part[8];
    #pragma unroll
    for (int j = 0; j < 8; ++j) {
        float h0 = c0[j] + b1lo; h0 = h0 > 0.f ? h0 : 0.f;
        float h1 = c1[j] + b1hi; h1 = h1 > 0.f ? h1 : 0.f;
        part[j] = h0 * w2lo + h1 * w2hi;
    }
    // reduce across the 16 lanes sharing `hi` (xor masks <= 8 stay in group)
    #pragma unroll
    for (int off = 1; off < 16; off <<= 1) {
        #pragma unroll
        for (int j = 0; j < 8; ++j)
            part[j] += __shfl_xor(part[j], off, 32);
    }
    if (n == 0) {
        const float b2 = b2p[0];
        #pragma unroll
        for (int j = 0; j < 8; ++j)
            scores[R0 + (hi << 3) + j] = part[j] + b2;
    }
}

// ---------------------------------------------------------------------------
// Kernel 2: p_gen gate + masked softmax over the K pointer scores.
// One block per batch row.
// ---------------------------------------------------------------------------
__global__ __launch_bounds__(256)
void gate_and_pointer_softmax(const float* __restrict__ dec,
                              const float* __restrict__ Wg1,  // [D,H]
                              const float* __restrict__ bg1,  // [H]
                              const float* __restrict__ Wg2,  // [H]
                              const float* __restrict__ bg2p, // [1]
                              const int*   __restrict__ vmask,// [B,K]
                              const float* __restrict__ scores,
                              float* __restrict__ pprobs,     // [B*K]
                              float* __restrict__ pgen,       // [B]
                              float* __restrict__ mass)       // [B]
{
    const int b = blockIdx.x;
    const int t = threadIdx.x;
    __shared__ float red[256];
    __shared__ float sh_max, sh_sum, sh_vsum;

    // ---- p_gen on the first wave (HH == 32 == wave width)
    if (t < HH) {
        float s = bg1[t];
        const float* dr = dec + (long)b * DD;
        for (int i = 0; i < DD; ++i) s += dr[i] * Wg1[i * HH + t];
        s = s > 0.f ? s : 0.f;
        float c = s * Wg2[t];
        #pragma unroll
        for (int off = 1; off < HH; off <<= 1) c += __shfl_xor(c, off, 32);
        if (t == 0) pgen[b] = 1.f / (1.f + expf(-(c + bg2p[0])));
    }

    // ---- masked softmax over K
    float lmax = -3.0e38f;
    for (int k = t; k < KK; k += blockDim.x) {
        const bool mv = vmask[b * KK + k] != 0;
        const float s = mv ? scores[b * KK + k] : -1e9f;
        lmax = s > lmax ? s : lmax;
    }
    red[t] = lmax; __syncthreads();
    for (int off = 128; off > 0; off >>= 1) {
        if (t < off) red[t] = red[t] > red[t + off] ? red[t] : red[t + off];
        __syncthreads();
    }
    if (t == 0) sh_max = red[0];
    __syncthreads();
    const float mx = sh_max;

    float lsum = 0.f, lvsum = 0.f;
    for (int k = t; k < KK; k += blockDim.x) {
        const bool mv = vmask[b * KK + k] != 0;
        const float s = mv ? scores[b * KK + k] : -1e9f;
        const float e = expf(s - mx);
        lsum += e;
        if (mv) lvsum += e;
    }
    red[t] = lsum; __syncthreads();
    for (int off = 128; off > 0; off >>= 1) {
        if (t < off) red[t] += red[t + off];
        __syncthreads();
    }
    if (t == 0) sh_sum = red[0];
    __syncthreads();
    red[t] = lvsum; __syncthreads();
    for (int off = 128; off > 0; off >>= 1) {
        if (t < off) red[t] += red[t + off];
        __syncthreads();
    }
    if (t == 0) sh_vsum = red[0];
    __syncthreads();

    const float inv = 1.f / sh_sum;
    for (int k = t; k < KK; k += blockDim.x) {
        const bool mv = vmask[b * KK + k] != 0;
        const float s = mv ? scores[b * KK + k] : -1e9f;
        pprobs[b * KK + k] = mv ? expf(s - mx) * inv : 0.f;
    }
    if (t == 0) mass[b] = sh_vsum * inv;
}

// ---------------------------------------------------------------------------
// Kernel 3: per-row vocab softmax stats (max, sum-exp). One block per row.
// ---------------------------------------------------------------------------
__global__ __launch_bounds__(256)
void vocab_stats(const float* __restrict__ logits,
                 float* __restrict__ rowmax, float* __restrict__ rowsum)
{
    const int b = blockIdx.x;
    const int t = threadIdx.x;
    __shared__ float red[256];
    __shared__ float shm;
    const float* lr = logits + (long)b * VV;

    float lmax = -3.0e38f;
    for (int v = t; v < VV; v += 256) { float x = lr[v]; lmax = x > lmax ? x : lmax; }
    red[t] = lmax; __syncthreads();
    for (int off = 128; off > 0; off >>= 1) {
        if (t < off) red[t] = red[t] > red[t + off] ? red[t] : red[t + off];
        __syncthreads();
    }
    if (t == 0) shm = red[0];
    __syncthreads();
    const float mx = shm;

    float lsum = 0.f;
    for (int v = t; v < VV; v += 256) lsum += expf(lr[v] - mx);
    red[t] = lsum; __syncthreads();
    for (int off = 128; off > 0; off >>= 1) {
        if (t < off) red[t] += red[t + off];
        __syncthreads();
    }
    if (t == 0) { rowmax[b] = mx; rowsum[b] = red[0]; }
}

// ---------------------------------------------------------------------------
// Kernel 4: dense output  out[b,v] = log(g * softmax(logits)[b,v] + eps)
//   g = 1 if pointer_mass < 0.5, else (1 - p_gen).   (pointer part patched later)
// ---------------------------------------------------------------------------
__global__ __launch_bounds__(256)
void write_out(const float* __restrict__ logits,
               const float* __restrict__ rowmax, const float* __restrict__ rowsum,
               const float* __restrict__ pgen,   const float* __restrict__ mass,
               float* __restrict__ out)
{
    const int b = blockIdx.y;
    const float mx  = rowmax[b];
    const float inv = 1.f / rowsum[b];
    const float g   = (mass[b] < THRESH) ? 1.f : (1.f - pgen[b]);
    const long base = (long)b * VV;

    const int chunk = (VV + gridDim.x - 1) / gridDim.x;
    const int v0 = blockIdx.x * chunk;
    int v1 = v0 + chunk; if (v1 > VV) v1 = VV;

    for (int v = v0 + (int)threadIdx.x; v < v1; v += blockDim.x) {
        int pv = v + 4096; if (pv >= VV) pv = VV - 1;          // stay in-row
        __builtin_prefetch(&logits[base + pv], 0, 1);          // global_prefetch_b8
        const float p = expf(logits[base + v] - mx) * inv;
        out[base + v] = logf(g * p + EPSV);
    }
}

// ---------------------------------------------------------------------------
// Kernel 5: patch the <=K scattered vocab entries per row.
// Duplicate token ids are merged in LDS (O(K^2) = 40K ops/row); all duplicate
// writers compute the identical value, so concurrent stores are benign.
// ---------------------------------------------------------------------------
__global__ __launch_bounds__(256)
void scatter_fix(const float* __restrict__ logits,
                 const int*   __restrict__ tokids, const float* __restrict__ pprobs,
                 const float* __restrict__ rowmax, const float* __restrict__ rowsum,
                 const float* __restrict__ pgen,   const float* __restrict__ mass,
                 float* __restrict__ out)
{
    const int b = blockIdx.x;
    const int t = threadIdx.x;
    __shared__ int   stok[KK];
    __shared__ float spp[KK];
    for (int k = t; k < KK; k += blockDim.x) {
        stok[k] = tokids[b * KK + k];
        spp[k]  = pprobs[b * KK + k];
    }
    __syncthreads();

    const float mx  = rowmax[b];
    const float inv = 1.f / rowsum[b];
    const float ms  = mass[b];
    const float pg  = pgen[b];

    for (int k = t; k < KK; k += blockDim.x) {
        const int tok = stok[k];
        float sum = 0.f;
        for (int j = 0; j < KK; ++j)
            if (stok[j] == tok) sum += spp[j];
        const float orig = expf(logits[(long)b * VV + tok] - mx) * inv;
        float val;
        if (ms < THRESH) val = logf(orig + EPSV);
        else             val = logf(pg * sum + (1.f - pg) * orig + EPSV);
        out[(long)b * VV + tok] = val;
    }
}

// ---------------------------------------------------------------------------
extern "C" void kernel_launch(void* const* d_in, const int* in_sizes, int n_in,
                              void* d_out, int out_size, void* d_ws, size_t ws_size,
                              hipStream_t stream)
{
    const float* logits = (const float*)d_in[0];
    const float* dec    = (const float*)d_in[1];
    const float* bemb   = (const float*)d_in[2];
    const float* W1     = (const float*)d_in[3];
    const float* b1     = (const float*)d_in[4];
    const float* W2     = (const float*)d_in[5];
    const float* b2     = (const float*)d_in[6];
    const float* Wg1    = (const float*)d_in[7];
    const float* bg1    = (const float*)d_in[8];
    const float* Wg2    = (const float*)d_in[9];
    const float* bg2    = (const float*)d_in[10];
    const int*   tokids = (const int*)d_in[11];
    const int*   subidx = (const int*)d_in[12];
    const int*   vmask  = (const int*)d_in[13];
    float* out = (float*)d_out;

    // Workspace layout (~413 KB)
    float* scores = (float*)d_ws;              // B*K
    float* pprobs = scores + BB * KK;          // B*K
    float* pgen   = pprobs + BB * KK;          // B
    float* mass   = pgen + BB;                 // B
    float* rowmax = mass + BB;                 // B
    float* rowsum = rowmax + BB;               // B

    // 1) pointer-MLP scores via WMMA: 51200 rows, 16 rows/wave, 8 waves/block
    mlp_scores_wmma<<<(BB * KK) / (16 * 8), 256, 0, stream>>>(
        dec, bemb, W1, b1, W2, b2, subidx, scores);

    // 2) p_gen gate + masked pointer softmax
    gate_and_pointer_softmax<<<BB, 256, 0, stream>>>(
        dec, Wg1, bg1, Wg2, bg2, vmask, scores, pprobs, pgen, mass);

    // 3) vocab softmax stats
    vocab_stats<<<BB, 256, 0, stream>>>(logits, rowmax, rowsum);

    // 4) dense log-prob output (pointer contribution zero everywhere)
    write_out<<<dim3(32, BB), 256, 0, stream>>>(
        logits, rowmax, rowsum, pgen, mass, out);

    // 5) patch the scattered pointer tokens
    scatter_fix<<<BB, 256, 0, stream>>>(
        logits, tokids, pprobs, rowmax, rowsum, pgen, mass, out);
}
